// MultiHeadAttention_54494545051965
// MI455X (gfx1250) — matrix operand order repaired and checked by reading, not compiled, not run on hardware
//
#include <hip/hip_runtime.h>
#include <hip/hip_bf16.h>

typedef __attribute__((ext_vector_type(16))) __bf16 v16bf;
typedef __attribute__((ext_vector_type(8)))  __bf16 v8bf;
typedef __attribute__((ext_vector_type(8)))  float  v8f;
typedef __attribute__((ext_vector_type(4)))  int    v4i;

#define WMMA_BF16(a,b,c) \
  __builtin_amdgcn_wmma_f32_16x16x32_bf16(false,(a),false,(b),(short)0,(c),false,false)

static constexpr int Bz = 4, Ss = 2048, Dd = 768, Hh = 12, DKk = 64;

#if defined(__HIP_DEVICE_COMPILE__) && defined(__has_builtin)
#if __has_builtin(__builtin_amdgcn_global_load_async_to_lds_b128)
#define HAVE_ASYNC_LDS 1
#endif
#endif
#ifndef HAVE_ASYNC_LDS
#define HAVE_ASYNC_LDS 0
#endif

__device__ __forceinline__ void copy16_to_lds(const __bf16* g, __bf16* l) {
#if HAVE_ASYNC_LDS
  __builtin_amdgcn_global_load_async_to_lds_b128(
      (__attribute__((address_space(1))) v4i*)g,
      (__attribute__((address_space(3))) v4i*)l, 0, 0);
#else
  *(v8bf*)l = *(const v8bf*)g;
#endif
}

__device__ __forceinline__ void wait_async_lds() {
#if HAVE_ASYNC_LDS
  asm volatile("s_wait_asynccnt 0" ::: "memory");
#endif
}

// combine two v8bf halves into a v16bf fragment
__device__ __forceinline__ v16bf frag16(v8bf lo8, v8bf hi8) {
  v16bf f;
#pragma unroll
  for (int e = 0; e < 8; ++e) { f[e] = lo8[e]; f[8 + e] = hi8[e]; }
  return f;
}

// ---------------------------------------------------------------------------
// Elementwise f32 -> bf16 (weight pre-conversion; amortizes B-frag converts)
// ---------------------------------------------------------------------------
__global__ __launch_bounds__(256) void cvt_bf16_kernel(const float* __restrict__ in,
                                                       __bf16* __restrict__ out, int n) {
  const int i = (blockIdx.x * 256 + threadIdx.x) * 4;
  if (i + 3 < n) {
    const float4 f = *(const float4*)(in + i);
    out[i + 0] = (__bf16)f.x;
    out[i + 1] = (__bf16)f.y;
    out[i + 2] = (__bf16)f.z;
    out[i + 3] = (__bf16)f.w;
  }
}

// ---------------------------------------------------------------------------
// GEMM: C[m,n] = sum_k X[m,k] * Wbf[n,k] + bias[n]   (x @ W^T + b), W in bf16
// MODE 0: bf16 -> [B,H,S,DK]; MODE 1: bf16 -> [B,H,DK,S]; MODE 2: f32 -> [M,N]
// XB: X already bf16. Wave tile 32x64 (8 WMMA/k-step); block 8 waves -> 256x64.
// ---------------------------------------------------------------------------
template <int MODE, bool XB>
__global__ __launch_bounds__(256) void gemm_xwT(const void* __restrict__ Xv,
                                                const __bf16* __restrict__ Wbf,
                                                const float* __restrict__ bias,
                                                void* __restrict__ out) {
  const int lane = threadIdx.x & 31;
  const int wave = threadIdx.x >> 5;
  const int lo = lane & 15, hi = lane >> 4;
  const int m0 = blockIdx.x * 256 + wave * 32;
  const int n0 = blockIdx.y * 64;
  const int K = Dd;

  v8f acc[2][4] = {};

  for (int k0 = 0; k0 < K; k0 += 32) {
    // A fragments for rows m0..+15 and m0+16..+31
    v16bf a[2];
#pragma unroll
    for (int u = 0; u < 2; ++u) {
      const int row = m0 + u * 16 + lo;
      const int bk = k0 + hi * 8;
      if (XB) {
        const v8bf* p = (const v8bf*)((const __bf16*)Xv + (size_t)row * K + bk);
        a[u] = frag16(p[0], p[2]);
      } else {
        const float* p = (const float*)Xv + (size_t)row * K + bk;
        __builtin_prefetch(p + 32, 0, 3);
#pragma unroll
        for (int e = 0; e < 8; ++e) { a[u][e] = (__bf16)p[e]; a[u][8 + e] = (__bf16)p[16 + e]; }
      }
    }
    // batch-issue all 4 B fragments, then run the 8-WMMA chain
    v16bf bf[4];
#pragma unroll
    for (int t = 0; t < 4; ++t) {
      const __bf16* p = Wbf + (size_t)(n0 + t * 16 + lo) * K + k0 + hi * 8;
      bf[t] = frag16(*(const v8bf*)p, *(const v8bf*)(p + 16));
    }
#pragma unroll
    for (int t = 0; t < 4; ++t)
#pragma unroll
      for (int u = 0; u < 2; ++u) acc[u][t] = WMMA_BF16(a[u], bf[t], acc[u][t]);
  }

  // Epilogue: C layout -> lane lo = N col, vgpr r -> M row (r + hi*8)
#pragma unroll
  for (int t = 0; t < 4; ++t) {
    const int n = n0 + t * 16 + lo;
    const float bn = bias[n];
#pragma unroll
    for (int u = 0; u < 2; ++u) {
#pragma unroll
      for (int r = 0; r < 8; ++r) {
        const int m = m0 + u * 16 + hi * 8 + r;
        const float val = acc[u][t][r] + bn;
        if (MODE == 2) {
          ((float*)out)[(size_t)m * Dd + n] = val;
        } else {
          const int b = m >> 11, s = m & (Ss - 1);
          const int h = n >> 6, dk = n & 63;
          const size_t idx = (MODE == 0)
              ? ((size_t)(b * Hh + h) * Ss + s) * DKk + dk
              : ((size_t)(b * Hh + h) * DKk + dk) * Ss + s;
          ((__bf16*)out)[idx] = (__bf16)val;
        }
      }
    }
  }
}

// ---------------------------------------------------------------------------
// Flash attention, workgroup-cooperative K/V staging in LDS (double-buffered).
// Q,K: bf16 [B,H,S,DK]; VT: bf16 [B,H,DK,S]; O: bf16 [B,S,D] (concat layout).
// Block = 8 waves x 16 q-rows = 128 q-rows per (b,h); key blocks of 32.
// ---------------------------------------------------------------------------
__global__ __launch_bounds__(256) void attn_kernel(const __bf16* __restrict__ Q,
                                                   const __bf16* __restrict__ Km,
                                                   const __bf16* __restrict__ VT,
                                                   const int* __restrict__ mask,
                                                   __bf16* __restrict__ O) {
  __shared__ __bf16 kbuf[2][32 * 64];   // [key][dk]   4KB x2
  __shared__ __bf16 vbuf[2][64 * 32];   // [dk][key]   4KB x2
  __shared__ __bf16 pbuf[8][16 * 32];   // per-wave P staging 8KB

  const int tid = threadIdx.x;
  const int lane = tid & 31;
  const int wave = tid >> 5;
  const int lo = lane & 15, hi = lane >> 4;
  const int bh = blockIdx.y;
  const int b = bh / Hh, h = bh % Hh;
  const int q0 = blockIdx.x * 128 + wave * 16;

  const __bf16* Qp = Q + (size_t)bh * Ss * DKk;
  const __bf16* Kp = Km + (size_t)bh * Ss * DKk;
  const __bf16* Vp = VT + (size_t)bh * DKk * Ss;
  __bf16* ldsw = pbuf[wave];

  // cooperative tile loader: 256 threads x 16B per tile
  auto load_tile = [&](int j0, int buf) {
    const __bf16* gk = Kp + (size_t)(j0 + (tid >> 3)) * DKk + (tid & 7) * 8;
    copy16_to_lds(gk, &kbuf[buf][(tid >> 3) * 64 + (tid & 7) * 8]);
    const __bf16* gv = Vp + (size_t)(tid >> 2) * Ss + j0 + (tid & 3) * 8;
    copy16_to_lds(gv, &vbuf[buf][(tid >> 2) * 32 + (tid & 3) * 8]);
  };

  // Q A-fragments for k0 = 0, 32 (resident)
  v16bf aq[2];
#pragma unroll
  for (int kk = 0; kk < 2; ++kk) {
    const __bf16* p = Qp + (size_t)(q0 + lo) * DKk + kk * 32 + hi * 8;
    aq[kk] = frag16(*(const v8bf*)p, *(const v8bf*)(p + 16));
  }

  bool mon[8];
  {
    const int* mp = mask + b * Ss + q0 + hi * 8;
#pragma unroll
    for (int r = 0; r < 8; ++r) mon[r] = (mp[r] != 0);
  }

  float mi[8], li[8];
#pragma unroll
  for (int r = 0; r < 8; ++r) { mi[r] = -3.0e38f; li[r] = 0.0f; }
  v8f o[4] = {};
  const float scale = 0.125f;  // 1/sqrt(64)

  load_tile(0, 0);
  wait_async_lds();
  __syncthreads();

  for (int j0 = 0; j0 < Ss; j0 += 32) {
    const int cur = (j0 >> 5) & 1;
    if (j0 + 32 < Ss) load_tile(j0 + 32, cur ^ 1);

    // --- batch-issue ALL fragment loads from LDS (K: 4, V: 4) ---
    v16bf bk[2][2];  // [jt][kk]
#pragma unroll
    for (int jt = 0; jt < 2; ++jt)
#pragma unroll
      for (int kk = 0; kk < 2; ++kk) {
        const __bf16* p = &kbuf[cur][(jt * 16 + lo) * 64 + kk * 32 + hi * 8];
        bk[jt][kk] = frag16(*(const v8bf*)p, *(const v8bf*)(p + 16));
      }
    v16bf bv[4];
#pragma unroll
    for (int t = 0; t < 4; ++t) {
      const __bf16* p = &vbuf[cur][(t * 16 + lo) * 32 + hi * 8];
      bv[t] = frag16(*(const v8bf*)p, *(const v8bf*)(p + 16));
    }

    // --- scores: two 16x16 tiles over keys [j0, j0+32), K-dim = 64 ---
    v8f s[2];
#pragma unroll
    for (int jt = 0; jt < 2; ++jt) {
      v8f z = {};
      z = WMMA_BF16(aq[0], bk[jt][0], z);
      z = WMMA_BF16(aq[1], bk[jt][1], z);
      s[jt] = z;
    }

    // --- online softmax (row reductions across 16 lanes of each half-wave) ---
    float p0[8], p1[8], corr[8];
#pragma unroll
    for (int r = 0; r < 8; ++r) {
      float a0 = mon[r] ? s[0][r] * scale : -1.0e9f;
      float a1 = mon[r] ? s[1][r] * scale : -1.0e9f;
      float rm = fmaxf(a0, a1);
      rm = fmaxf(rm, __shfl_xor(rm, 1, 32));
      rm = fmaxf(rm, __shfl_xor(rm, 2, 32));
      rm = fmaxf(rm, __shfl_xor(rm, 4, 32));
      rm = fmaxf(rm, __shfl_xor(rm, 8, 32));
      const float mn = fmaxf(mi[r], rm);
      corr[r] = __expf(mi[r] - mn);
      p0[r] = __expf(a0 - mn);
      p1[r] = __expf(a1 - mn);
      float rs = p0[r] + p1[r];
      rs += __shfl_xor(rs, 1, 32);
      rs += __shfl_xor(rs, 2, 32);
      rs += __shfl_xor(rs, 4, 32);
      rs += __shfl_xor(rs, 8, 32);
      li[r] = li[r] * corr[r] + rs;
      mi[r] = mn;
    }
#pragma unroll
    for (int t = 0; t < 4; ++t)
#pragma unroll
      for (int r = 0; r < 8; ++r) o[t][r] *= corr[r];

    // --- P (C layout) -> per-wave LDS -> A layout bf16 ---
#pragma unroll
    for (int r = 0; r < 8; ++r) {
      const int row = hi * 8 + r;
      ldsw[row * 32 + lo] = (__bf16)p0[r];
      ldsw[row * 32 + 16 + lo] = (__bf16)p1[r];
    }
    v16bf ap;
    {
      const int bkk = hi * 8;
      ap = frag16(*(const v8bf*)&ldsw[lo * 32 + bkk],
                  *(const v8bf*)&ldsw[lo * 32 + bkk + 16]);
    }

    // --- PV: V fragments already resident in registers ---
#pragma unroll
    for (int t = 0; t < 4; ++t) o[t] = WMMA_BF16(ap, bv[t], o[t]);

    wait_async_lds();   // next tile landed
    __syncthreads();    // all waves done with cur, next tile visible
  }

  // epilogue: normalize and store concat layout [b, s, h*64 + dk] as bf16
#pragma unroll
  for (int r = 0; r < 8; ++r) {
    const float inv = 1.0f / li[r];
    const int qrow = q0 + hi * 8 + r;
#pragma unroll
    for (int t = 0; t < 4; ++t) {
      O[((size_t)(b * Ss + qrow)) * Dd + h * 64 + t * 16 + lo] = (__bf16)(o[t][r] * inv);
    }
  }
}

// ---------------------------------------------------------------------------
extern "C" void kernel_launch(void* const* d_in, const int* in_sizes, int n_in,
                              void* d_out, int out_size, void* d_ws, size_t ws_size,
                              hipStream_t stream) {
  const float* q   = (const float*)d_in[0];
  const float* k   = (const float*)d_in[1];
  const float* v   = (const float*)d_in[2];
  const int*   msk = (const int*)d_in[3];
  const float* Wq  = (const float*)d_in[4];
  const float* bq  = (const float*)d_in[5];
  const float* Wk  = (const float*)d_in[6];
  const float* bk  = (const float*)d_in[7];
  const float* Wv  = (const float*)d_in[8];
  const float* bv  = (const float*)d_in[9];
  const float* Wo  = (const float*)d_in[10];
  const float* bo  = (const float*)d_in[11];

  const size_t elems = (size_t)Bz * Ss * Dd;      // 6.29M per activation tensor
  const int    wlen  = Dd * Dd;                   // 589824 per weight
  __bf16* wsQ = (__bf16*)d_ws;
  __bf16* wsK = wsQ + elems;
  __bf16* wsV = wsK + elems;
  __bf16* wsO = wsV + elems;
  __bf16* wsW = wsO + elems;                      // single reusable weight buffer

  dim3 blk(256);
  dim3 gg(32, 12);                 // (8192/256) x (768/64)
  dim3 ga(16, 48);                 // (S/128) x (B*H)
  dim3 gc((wlen / 4 + 255) / 256); // weight conversion

  cvt_bf16_kernel<<<gc, blk, 0, stream>>>(Wq, wsW, wlen);
  gemm_xwT<0, false><<<gg, blk, 0, stream>>>(q, wsW, bq, wsQ);
  cvt_bf16_kernel<<<gc, blk, 0, stream>>>(Wk, wsW, wlen);
  gemm_xwT<0, false><<<gg, blk, 0, stream>>>(k, wsW, bk, wsK);
  cvt_bf16_kernel<<<gc, blk, 0, stream>>>(Wv, wsW, wlen);
  gemm_xwT<1, false><<<gg, blk, 0, stream>>>(v, wsW, bv, wsV);
  attn_kernel<<<ga, blk, 0, stream>>>(wsQ, wsK, wsV, msk, wsO);
  cvt_bf16_kernel<<<gc, blk, 0, stream>>>(Wo, wsW, wlen);
  gemm_xwT<2, true><<<gg, blk, 0, stream>>>(wsO, wsW, bo, d_out);
}
